// SplitCGCconv_19018115187172
// MI455X (gfx1250) — matrix operand order repaired and checked by reading, not compiled, not run on hardware
//
#include <hip/hip_runtime.h>
#include <hip/hip_bf16.h>
#include <math.h>

typedef __attribute__((ext_vector_type(16))) _Float16 v16h;
typedef __attribute__((ext_vector_type(8)))  float    v8f;

#define HWPIX 9216   // 96*96
#define WIMG  96

// ---------------- workspace layout (bytes) ----------------
static const size_t WS_OFF  = 0;                                      // off maps: 8*512*9216 f32
static const size_t WS_XH   = WS_OFF  + (size_t)8 * 512 * HWPIX * 4;  // x fp16:   8*256*9216 f16
static const size_t WS_XSH  = WS_XH   + (size_t)8 * 256 * HWPIX * 2;  // deformed+split fp16: 32*64*9216
static const size_t WS_GL   = WS_XSH  + (size_t)32 * 64 * HWPIX * 2;  // gl: 32*64*9 f32
static const size_t WS_GC   = WS_GL   + (size_t)32 * 64 * 9 * 4;      // gate_c
static const size_t WS_GO   = WS_GC   + (size_t)32 * 64 * 9 * 4;      // gate_o
static const size_t WS_WGT  = WS_GO   + (size_t)32 * 64 * 9 * 4;      // wgt fp16: 32*64*576
static const size_t WS_WOFH = WS_WGT  + (size_t)32 * 64 * 576 * 2;    // offset_w fp16: 512*2304

// ---------------- CDNA5 async global->LDS copy (ASYNCcnt) ----------------
__device__ __forceinline__ void async_copy_b128(const _Float16* gsrc, _Float16* ldst) {
  auto l = (__attribute__((address_space(3))) _Float16*)ldst;   // 32-bit LDS address
  auto g = (const __attribute__((address_space(1))) _Float16*)gsrc;
  asm volatile("global_load_async_to_lds_b128 %0, %1, off" :: "v"(l), "v"(g) : "memory");
}
__device__ __forceinline__ void wait_async0() {
  asm volatile("s_wait_asynccnt 0x0" ::: "memory");
}

// ---------------- 1) fp32 -> fp16 converts, 8 elems/thread ----------------
__global__ __launch_bounds__(256) void cvt_f16_kernel(const float* __restrict__ x,
                                                      _Float16* __restrict__ xh, int n) {
  int i = (blockIdx.x * 256 + threadIdx.x) * 8;
  if (i >= n) return;
  float4 a = *(const float4*)(x + i);
  float4 b = *(const float4*)(x + i + 4);
  __attribute__((aligned(16))) _Float16 t[8] = {
      (_Float16)a.x, (_Float16)a.y, (_Float16)a.z, (_Float16)a.w,
      (_Float16)b.x, (_Float16)b.y, (_Float16)b.z, (_Float16)b.w};
  *(float4*)(xh + i) = *(const float4*)&t[0];
}

// ---- persistent per-thread im2col walker (one n-column, consecutive k's) ----
// state: jw in [0,9), offw = offset of x[c][h0 + jw/3 - 1][w0 + jw%3 - 1] rel. to xb.
// Branchless: invalid (halo) elements load from a safe offset and select 0.
struct Im2colWalk {
  const _Float16* xb;   // uniform per block
  int offw;
  int jw;
  unsigned vmask;
  __device__ __forceinline__ void init(const _Float16* xb_, int h0, int w0) {
    xb = xb_;
    jw = 0;
    offw = (h0 - 1) * WIMG + (w0 - 1);
    vmask = 0;
#pragma unroll
    for (int jj = 0; jj < 9; ++jj) {
      int hh = h0 + jj / 3 - 1, wv = w0 + jj % 3 - 1;
      if ((unsigned)hh < 96u && (unsigned)wv < 96u) vmask |= 1u << jj;
    }
  }
  __device__ __forceinline__ _Float16 step() {
    bool valid = (vmask >> jw) & 1u;
    int o = valid ? offw : 0;           // safe offset, always in range
    _Float16 v = xb[o];                 // unconditional load (no exec branch)
    v = valid ? v : (_Float16)0.0f;     // cndmask
    int d = ((jw == 2) | (jw == 5)) ? 94 : 1;
    d = (jw == 8) ? (HWPIX - 194) : d;
    offw += d;
    jw = (jw == 8) ? 0 : jw + 1;
    return v;
  }
};

// ---------------- 2) offset conv: implicit GEMM via WMMA ----------------
// A = offset_w (512 x 2304) fp16, B = im2col(x_h) (2304 x 9216) per image.
// Block 128x256, K step 32, 8 waves (2x4), wave tile 64x64 (16 wmma).
// Double-buffered LDS (pipeline unrolled x2), A tile via async-to-LDS DMA.
__global__ __launch_bounds__(256) void offset_gemm_kernel(const _Float16* __restrict__ xh,
                                                          const _Float16* __restrict__ wOffH,
                                                          float* __restrict__ off) {
  __shared__ _Float16 As[2][128][40];
  __shared__ _Float16 Bs[2][256][40];

  const int b    = blockIdx.z;
  const int mBlk = blockIdx.y;          // 0..3
  const int nBlk = blockIdx.x;          // 0..35
  const int tid  = threadIdx.x;
  const int lane = tid & 31;
  const int wave = tid >> 5;
  const int waveM = wave >> 2;          // 0..1
  const int waveN = wave & 3;           // 0..3

  const _Float16* xb = xh + (size_t)b * 256 * HWPIX;

  const int nCol = tid;                 // one n-column per thread
  const int pix  = nBlk * 256 + nCol;
  const int h0   = pix / WIMG;
  const int w0   = pix - h0 * WIMG;

  Im2colWalk walk;
  walk.init(xb, h0, w0);

  auto loadA = [&](int p, int k0) {
#pragma unroll
    for (int j = 0; j < 2; ++j) {                 // 512 b128 chunks / 256 threads
      int chunk = tid + j * 256;
      int m = chunk >> 2, seg = (chunk & 3) * 8;
      async_copy_b128(wOffH + (size_t)(mBlk * 128 + m) * 2304 + k0 + seg, &As[p][m][seg]);
    }
  };
  auto loadB = [&](int p) {
    __attribute__((aligned(16))) _Float16 tmp[32];
#pragma unroll
    for (int t = 0; t < 32; ++t) tmp[t] = walk.step();
#pragma unroll
    for (int s = 0; s < 4; ++s)
      *(float4*)&Bs[p][nCol][s * 8] = *(const float4*)&tmp[s * 8];
  };

  v8f acc[4][4] = {};
  const int mA  = lane & 15;
  const int kbA = (lane < 16) ? 0 : 8;
  const int nB  = lane & 15;
  const int kbB = (lane < 16) ? 0 : 8;

  auto compute = [&](int p) {
    v16h afrag[4], bfrag[4];
#pragma unroll
    for (int mi = 0; mi < 4; ++mi) {
      const _Float16* q = &As[p][waveM * 64 + mi * 16 + mA][kbA];
#pragma unroll
      for (int t = 0; t < 8; ++t) { afrag[mi][t] = q[t]; afrag[mi][t + 8] = q[16 + t]; }
    }
#pragma unroll
    for (int ni = 0; ni < 4; ++ni) {
      const _Float16* q = &Bs[p][waveN * 64 + ni * 16 + nB][kbB];
#pragma unroll
      for (int t = 0; t < 8; ++t) { bfrag[ni][t] = q[t]; bfrag[ni][t + 8] = q[16 + t]; }
    }
#pragma unroll
    for (int mi = 0; mi < 4; ++mi)
#pragma unroll
      for (int ni = 0; ni < 4; ++ni)
        acc[mi][ni] = __builtin_amdgcn_wmma_f32_16x16x32_f16(
            false, afrag[mi], false, bfrag[ni], (short)0, acc[mi][ni], false, false);
  };

  loadA(0, 0);
  loadB(0);
  for (int k0 = 0; k0 < 2304; k0 += 64) {
    wait_async0();
    __syncthreads();
    if (k0 + 32 < 2304) { loadA(1, k0 + 32); loadB(1); }
    compute(0);
    wait_async0();
    __syncthreads();
    if (k0 + 64 < 2304) { loadA(0, k0 + 64); loadB(0); }
    compute(1);
  }

  float* outb = off + (size_t)b * 512 * HWPIX;
  const int mAdd = (lane < 16) ? 0 : 8;
  const int nL   = lane & 15;
#pragma unroll
  for (int mi = 0; mi < 4; ++mi)
#pragma unroll
    for (int ni = 0; ni < 4; ++ni) {
      int n = nBlk * 256 + waveN * 64 + ni * 16 + nL;
#pragma unroll
      for (int v = 0; v < 8; ++v) {
        int m = mBlk * 128 + waveM * 64 + mi * 16 + mAdd + v;
        outb[(size_t)m * HWPIX + n] = acc[mi][ni][v];
      }
    }
}

// ---------------- 3) deform (bilinear) + radix split into GEMM2 batch layout ----
__global__ __launch_bounds__(256) void deform_kernel(const float* __restrict__ x,
                                                     const float* __restrict__ off,
                                                     _Float16* __restrict__ xsh) {
  size_t i = (size_t)blockIdx.x * 256 + threadIdx.x;   // over 8*256*9216
  if (i >= (size_t)8 * 256 * HWPIX) return;
  int pix = (int)(i % HWPIX);
  int bc  = (int)(i / HWPIX);
  int c   = bc & 255, b = bc >> 8;
  int gy = pix / WIMG, gx = pix - (pix / WIMG) * WIMG;
  const float* op = off + ((size_t)bc * HWPIX + pix) * 2;  // row-major reinterpretation
  float cy = fminf(fmaxf(op[0] + (float)gy, 0.0f), 95.0f);
  float cx = fminf(fmaxf(op[1] + (float)gx, 0.0f), 95.0f);
  float y0f = floorf(cy), x0f = floorf(cx);
  float y1f = ceilf(cy),  x1f = ceilf(cx);
  float fy = cy - y0f, fx = cx - x0f;
  int y0 = (int)y0f, x0 = (int)x0f, y1 = (int)y1f, x1 = (int)x1f;
  const float* xp = x + (size_t)bc * HWPIX;
  float v00 = xp[y0 * WIMG + x0], v01 = xp[y0 * WIMG + x1];
  float v10 = xp[y1 * WIMG + x0], v11 = xp[y1 * WIMG + x1];
  float top = v00 + fx * (v01 - v00);
  float bot = v10 + fx * (v11 - v10);
  float val = top + fy * (bot - top);
  size_t o = ((size_t)(((c >> 6) * 8 + b) * 64 + (c & 63))) * HWPIX + pix;
  xsh[o] = (_Float16)val;
}

// ---------------- 4) adaptive 3x3 avg pool over xs ----------------
__global__ __launch_bounds__(96) void pool_kernel(const _Float16* __restrict__ xsh,
                                                  float* __restrict__ gl) {
  __shared__ float sb[9];
  int bc = blockIdx.x;
  if (threadIdx.x < 9) sb[threadIdx.x] = 0.0f;
  __syncthreads();
  int r = threadIdx.x;
  const _Float16* p = xsh + (size_t)bc * HWPIX + r * WIMG;
  float s0 = 0.f, s1 = 0.f, s2 = 0.f;
#pragma unroll 8
  for (int w = 0; w < 32; ++w) { s0 += (float)p[w]; s1 += (float)p[32 + w]; s2 += (float)p[64 + w]; }
  int rb = (r >> 5) * 3;
  atomicAdd(&sb[rb + 0], s0);
  atomicAdd(&sb[rb + 1], s1);
  atomicAdd(&sb[rb + 2], s2);
  __syncthreads();
  if (threadIdx.x < 9) gl[(size_t)bc * 9 + threadIdx.x] = sb[threadIdx.x] * (1.0f / 1024.0f);
}

// ---------------- 5) gating MLPs: gate_c / gate_o ----------------
__global__ __launch_bounds__(64) void gate_kernel(const float* __restrict__ gl,
                                                  const float* __restrict__ ce_w,
                                                  const float* __restrict__ gd_w,
                                                  const float* __restrict__ ci_w,
                                                  const float* __restrict__ gd2_w,
                                                  const float* __restrict__ ce_g, const float* __restrict__ ce_b,
                                                  const float* __restrict__ ci2_g, const float* __restrict__ ci2_b,
                                                  const float* __restrict__ ci_g, const float* __restrict__ ci_b,
                                                  float* __restrict__ gate_c,
                                                  float* __restrict__ gate_o) {
  __shared__ float s_r2[64][5];
  const float inv = rsqrtf(1.0f + 1e-5f);
  int b = blockIdx.x, c = threadIdx.x;
  const float* g = gl + ((size_t)b * 64 + c) * 9;
  float gl9[9];
#pragma unroll
  for (int k = 0; k < 9; ++k) gl9[k] = g[k];
  float ce2[5];
#pragma unroll
  for (int l = 0; l < 5; ++l) {
    float s = 0.f;
#pragma unroll
    for (int k = 0; k < 9; ++k) s += gl9[k] * ce_w[l * 9 + k];
    ce2[l] = s;
  }
  {
    float sc = ce_g[c] * inv, bb = ce_b[c];
    float r[5];
#pragma unroll
    for (int l = 0; l < 5; ++l) r[l] = fmaxf(ce2[l] * sc + bb, 0.0f);
#pragma unroll
    for (int k = 0; k < 9; ++k) {
      float s = 0.f;
#pragma unroll
      for (int l = 0; l < 5; ++l) s += r[l] * gd_w[k * 5 + l];
      gate_c[((size_t)b * 64 + c) * 9 + k] = s;
    }
  }
  {
    float sc = ci2_g[c] * inv, bb = ci2_b[c];
#pragma unroll
    for (int l = 0; l < 5; ++l) s_r2[c][l] = fmaxf(ce2[l] * sc + bb, 0.0f);
  }
  __syncthreads();
  {
    int grp = c >> 4, o = c & 15;
    float t2[5];
#pragma unroll
    for (int l = 0; l < 5; ++l) {
      float s = 0.f;
#pragma unroll
      for (int i = 0; i < 16; ++i) s += s_r2[grp * 16 + i][l] * ci_w[o * 16 + i];
      t2[l] = s;
    }
    float sc = ci_g[c] * inv, bb = ci_b[c];
    float r[5];
#pragma unroll
    for (int l = 0; l < 5; ++l) r[l] = fmaxf(t2[l] * sc + bb, 0.0f);
#pragma unroll
    for (int k = 0; k < 9; ++k) {
      float s = 0.f;
#pragma unroll
      for (int l = 0; l < 5; ++l) s += r[l] * gd2_w[k * 5 + l];
      gate_o[((size_t)b * 64 + c) * 9 + k] = s;
    }
  }
}

// ---------------- 6) per-batch gated weights (fp16) ----------------
__global__ __launch_bounds__(256) void wgt_kernel(const float* __restrict__ gate_c,
                                                  const float* __restrict__ gate_o,
                                                  const float* __restrict__ w,
                                                  _Float16* __restrict__ wgt) {
  int i = blockIdx.x * 256 + threadIdx.x;             // over 32*64*576
  if (i >= 32 * 64 * 576) return;
  int k = i % 576;
  int bo = i / 576;
  int o = bo & 63, b = bo >> 6;
  int c = k / 9, j = k - c * 9;
  float gsum = gate_c[((size_t)b * 64 + c) * 9 + j] + gate_o[((size_t)b * 64 + o) * 9 + j];
  float gate = 1.0f / (1.0f + expf(-gsum));
  wgt[i] = (_Float16)(gate * w[(size_t)o * 576 + k]);
}

// ---------------- 7) dynamic-weight conv: batched implicit GEMM via WMMA ------
// per batch b2: A = wgt[b2] (64 x 576), B = im2col(xs_h[b2]) (576 x 9216)
// Block 64x256, K step 32, 8 waves (1x8), wave tile 64x32, double-buffered LDS.
__global__ __launch_bounds__(256) void cgc_gemm_kernel(const _Float16* __restrict__ xsh,
                                                       const _Float16* __restrict__ wgt,
                                                       float* __restrict__ out) {
  __shared__ _Float16 As[2][64][40];
  __shared__ _Float16 Bs[2][256][40];

  const int b2   = blockIdx.z;
  const int nBlk = blockIdx.x;          // 0..35
  const int tid  = threadIdx.x;
  const int lane = tid & 31;
  const int waveN = tid >> 5;           // 0..7

  const _Float16* xb = xsh + (size_t)b2 * 64 * HWPIX;
  const _Float16* ab = wgt + (size_t)b2 * 64 * 576;

  const int nCol = tid;
  const int pix  = nBlk * 256 + nCol;
  const int h0   = pix / WIMG;
  const int w0   = pix - h0 * WIMG;

  Im2colWalk walk;
  walk.init(xb, h0, w0);

  auto loadA = [&](int p, int k0) {                  // 256 b128 chunks / 256 threads
    int m = tid >> 2, seg = (tid & 3) * 8;
    async_copy_b128(ab + (size_t)m * 576 + k0 + seg, &As[p][m][seg]);
  };
  auto loadB = [&](int p) {
    __attribute__((aligned(16))) _Float16 tmp[32];
#pragma unroll
    for (int t = 0; t < 32; ++t) tmp[t] = walk.step();
#pragma unroll
    for (int s = 0; s < 4; ++s)
      *(float4*)&Bs[p][nCol][s * 8] = *(const float4*)&tmp[s * 8];
  };

  v8f acc[4][2] = {};
  const int mA  = lane & 15;
  const int kbA = (lane < 16) ? 0 : 8;
  const int nB  = lane & 15;
  const int kbB = (lane < 16) ? 0 : 8;

  auto compute = [&](int p) {
    v16h afrag[4], bfrag[2];
#pragma unroll
    for (int mi = 0; mi < 4; ++mi) {
      const _Float16* q = &As[p][mi * 16 + mA][kbA];
#pragma unroll
      for (int t = 0; t < 8; ++t) { afrag[mi][t] = q[t]; afrag[mi][t + 8] = q[16 + t]; }
    }
#pragma unroll
    for (int ni = 0; ni < 2; ++ni) {
      const _Float16* q = &Bs[p][waveN * 32 + ni * 16 + nB][kbB];
#pragma unroll
      for (int t = 0; t < 8; ++t) { bfrag[ni][t] = q[t]; bfrag[ni][t + 8] = q[16 + t]; }
    }
#pragma unroll
    for (int mi = 0; mi < 4; ++mi)
#pragma unroll
      for (int ni = 0; ni < 2; ++ni)
        acc[mi][ni] = __builtin_amdgcn_wmma_f32_16x16x32_f16(
            false, afrag[mi], false, bfrag[ni], (short)0, acc[mi][ni], false, false);
  };

  loadA(0, 0);
  loadB(0);
  for (int k0 = 0; k0 < 576; k0 += 64) {
    wait_async0();
    __syncthreads();
    if (k0 + 32 < 576) { loadA(1, k0 + 32); loadB(1); }
    compute(0);
    wait_async0();
    __syncthreads();
    if (k0 + 64 < 576) { loadA(0, k0 + 64); loadB(0); }
    compute(1);
  }

  const int borig = b2 & 7, r = b2 >> 3;
  float* ob = out + ((size_t)borig * 256 + r * 64) * HWPIX;
  const int mAdd = (lane < 16) ? 0 : 8;
  const int nL   = lane & 15;
#pragma unroll
  for (int mi = 0; mi < 4; ++mi)
#pragma unroll
    for (int ni = 0; ni < 2; ++ni) {
      int n = nBlk * 256 + waveN * 32 + ni * 16 + nL;
#pragma unroll
      for (int v = 0; v < 8; ++v) {
        int m = mi * 16 + mAdd + v;
        ob[(size_t)m * HWPIX + n] = acc[mi][ni][v];
      }
    }
}

// ---------------- launch ----------------
extern "C" void kernel_launch(void* const* d_in, const int* in_sizes, int n_in,
                              void* d_out, int out_size, void* d_ws, size_t ws_size,
                              hipStream_t stream) {
  const float* x        = (const float*)d_in[0];
  const float* offset_w = (const float*)d_in[1];
  const float* w        = (const float*)d_in[2];
  const float* ce_w     = (const float*)d_in[3];
  const float* gd_w     = (const float*)d_in[4];
  const float* ci_w     = (const float*)d_in[5];
  const float* gd2_w    = (const float*)d_in[6];
  const float* ce_g     = (const float*)d_in[7];
  const float* ce_b     = (const float*)d_in[8];
  const float* ci2_g    = (const float*)d_in[9];
  const float* ci2_b    = (const float*)d_in[10];
  const float* ci_g     = (const float*)d_in[11];
  const float* ci_b     = (const float*)d_in[12];
  float* out = (float*)d_out;

  char* ws = (char*)d_ws;
  float*    off_buf = (float*)(ws + WS_OFF);
  _Float16* xh      = (_Float16*)(ws + WS_XH);
  _Float16* xsh     = (_Float16*)(ws + WS_XSH);
  float*    gl      = (float*)(ws + WS_GL);
  float*    gate_c  = (float*)(ws + WS_GC);
  float*    gate_o  = (float*)(ws + WS_GO);
  _Float16* wgt     = (_Float16*)(ws + WS_WGT);
  _Float16* wOffH   = (_Float16*)(ws + WS_WOFH);

  const int nElems = 8 * 256 * HWPIX;     // 18,874,368
  const int nWOff  = 512 * 2304;          // 1,179,648

  cvt_f16_kernel<<<nElems / 8 / 256, 256, 0, stream>>>(x, xh, nElems);
  cvt_f16_kernel<<<nWOff / 8 / 256, 256, 0, stream>>>(offset_w, wOffH, nWOff);
  offset_gemm_kernel<<<dim3(36, 4, 8), 256, 0, stream>>>(xh, wOffH, off_buf);
  deform_kernel<<<(nElems + 255) / 256, 256, 0, stream>>>(x, off_buf, xsh);
  pool_kernel<<<32 * 64, 96, 0, stream>>>(xsh, gl);
  gate_kernel<<<32, 64, 0, stream>>>(gl, ce_w, gd_w, ci_w, gd2_w,
                                     ce_g, ce_b, ci2_g, ci2_b, ci_g, ci_b,
                                     gate_c, gate_o);
  wgt_kernel<<<(32 * 64 * 576 + 255) / 256, 256, 0, stream>>>(gate_c, gate_o, w, wgt);
  cgc_gemm_kernel<<<dim3(36, 1, 32), 256, 0, stream>>>(xsh, wgt, out);
}
